// ResidualMamba2_44848048505418
// MI455X (gfx1250) — matrix operand order repaired and checked by reading, not compiled, not run on hardware
//
#include <hip/hip_runtime.h>
#include <hip/hip_bf16.h>
#include <cstdint>

// ---------------------------------------------------------------------------
// Mamba2 residual block for MI455X (gfx1250, wave32, WMMA).
// All matmuls use v_wmma_f32_16x16x32_bf16 (bf16 operands, fp32 accum).
// GEMM B-operand tiles are staged cooperatively into LDS with the CDNA5
// async-copy path (global_load_async_to_lds_b128 + s_wait_asynccnt),
// double-buffered to overlap copy(i+1) with WMMA(i).
// ---------------------------------------------------------------------------

typedef __attribute__((ext_vector_type(16))) __bf16 v16bf;
typedef __attribute__((ext_vector_type(8)))  __bf16 v8bf;
typedef __attribute__((ext_vector_type(8)))  float  v8f;

// Problem constants
constexpr int BATCH  = 2;
constexpr int LSEQ   = 2048;
constexpr int TTOT   = BATCH * LSEQ;      // 4096 tokens
constexpr int DM     = 2048;              // d_model
constexpr int DI     = 4096;              // d_inner
constexpr int NH     = 64;                // heads
constexpr int HD     = 64;                // head dim
constexpr int DSTATE = 128;
constexpr int CONVD  = DI + 2 * DSTATE;   // 4352
constexpr int DIP    = 2 * DI + 2 * DSTATE + NH; // 8512
constexpr int NC     = LSEQ / 256;        // 8 chunks per batch
constexpr int CK     = 256;               // chunk length
constexpr int DCONV  = 4;

static __device__ __forceinline__ float siluf(float x) {
    return x / (1.0f + __expf(-x));
}
static __device__ __forceinline__ __bf16 f2bf(float x) { return (__bf16)x; }

// --------------------------- WMMA fragment loads ---------------------------
// A operand (16x32 bf16, row-major source, K contiguous):
//   elements 0..7  = K base + (lane<16 ? 0 : 8) + i
//   elements 8..15 = K base + (lane<16 ? 16 : 24) + i
static __device__ __forceinline__ v16bf load_a_frag(const __bf16* rowk, int lane) {
    const int klo = (lane < 16) ? 0 : 8;
    union { v16bf v; v8bf h[2]; } u;
    u.h[0] = *(const v8bf*)(rowk + klo);
    u.h[1] = *(const v8bf*)(rowk + klo + 16);
    return u.v;
}
// B operand (32x16 bf16): column = lane%16, K order same as A fragment
static __device__ __forceinline__ v16bf load_b_frag(const __bf16* rowk, int lane) {
    const int klo = (lane < 16) ? 0 : 8;
    union { v16bf v; v8bf h[2]; } u;
    u.h[0] = *(const v8bf*)(rowk + klo);
    u.h[1] = *(const v8bf*)(rowk + klo + 16);
    return u.v;
}

// CDNA5 async copy: one 16-byte global->LDS transfer, tracked by ASYNCcnt.
static __device__ __forceinline__ void async_copy_b128(const void* gptr, void* lptr) {
    const unsigned lds = (unsigned)(uintptr_t)lptr;               // low 32 bits = LDS offset
    const unsigned long long ga = (unsigned long long)(uintptr_t)gptr;
    asm volatile("global_load_async_to_lds_b128 %0, %1, off"
                 :: "v"(lds), "v"(ga) : "memory");
}
static __device__ __forceinline__ void wait_async_le1() {
    asm volatile("s_wait_asynccnt 0x1" ::: "memory");
}
static __device__ __forceinline__ void wait_async_0() {
    asm volatile("s_wait_asynccnt 0x0" ::: "memory");
}

// ------------------------------ Generic GEMM -------------------------------
// C[M,N] (+= residual) = A[M,K](bf16,row-major) * B[N,K](bf16,"N-major,K-contig")
// blockDim = 256 (8 waves). Each wave: 16 rows x 64 cols. Block: 128 x 64.
// B tiles (64 n x 32 k) are async-staged into double-buffered LDS: each of
// the 256 threads issues one b128 async load per tile.
// All M,N,K used are exact multiples of 128/64/32 respectively.
template <int EPI>
__global__ __launch_bounds__(256)
void k_gemm(const __bf16* __restrict__ A, const __bf16* __restrict__ B,
            float* __restrict__ C, const float* __restrict__ R,
            int M, int N, int K, int lda, int ldb, int ldc,
            long strA, long strB, long strC)
{
    __shared__ __bf16 Bl[2][64][32];     // 2 x 4KB double-buffered B tile

    const long bz = blockIdx.z;
    A += bz * strA; B += bz * strB; C += bz * strC;
    const int lane = threadIdx.x & 31;
    const int wid  = threadIdx.x >> 5;
    const int m0   = blockIdx.y * 128 + wid * 16;
    const int n0   = blockIdx.x * 64;

    // this thread's async-copy slot: row n_c (0..63), 8-element k segment
    const int n_c = threadIdx.x >> 2;
    const int k_c = (threadIdx.x & 3) * 8;
    const __bf16* bsrc = B + (long)(n0 + n_c) * ldb + k_c;

    v8f acc[4];
#pragma unroll
    for (int nt = 0; nt < 4; ++nt) acc[nt] = (v8f){0,0,0,0,0,0,0,0};

    const __bf16* arow = A + (long)(m0 + (lane & 15)) * lda;
    const int ntiles = K >> 5;

    async_copy_b128(bsrc, &Bl[0][n_c][k_c]);                 // prologue: tile 0
    for (int i = 0; i < ntiles; ++i) {
        if (i + 1 < ntiles) {
            async_copy_b128(bsrc + (i + 1) * 32, &Bl[(i + 1) & 1][n_c][k_c]);
            wait_async_le1();                                // tile i has landed
        } else {
            wait_async_0();
        }
        __syncthreads();                                     // tile visible to all waves

        const int k = i * 32;
        if (i + 2 < ntiles) __builtin_prefetch(arow + (i + 2) * 32, 0, 0);
        v16bf a = load_a_frag(arow + k, lane);
#pragma unroll
        for (int nt = 0; nt < 4; ++nt) {
            const int nl = nt * 16 + (lane & 15);
            v16bf b = load_b_frag(&Bl[i & 1][nl][0], lane);
            acc[nt] = __builtin_amdgcn_wmma_f32_16x16x32_bf16(
                false, a, false, b, (short)0, acc[nt], false, false);
        }
        __syncthreads();                                     // protect buffer for tile i+2
    }

#pragma unroll
    for (int nt = 0; nt < 4; ++nt) {
#pragma unroll
        for (int r = 0; r < 8; ++r) {
            const int mm = m0 + r + ((lane < 16) ? 0 : 8);
            const int nn = n0 + nt * 16 + (lane & 15);
            float v = acc[nt][r];
            if (EPI == 1) v += R[(long)mm * ldc + nn];
            C[(long)mm * ldc + nn] = v;
        }
    }
}

// --------------------------- Elementwise kernels ---------------------------
__global__ void k_cvt_bf16(const float* __restrict__ in, __bf16* __restrict__ out, int n) {
    int i = blockIdx.x * blockDim.x + threadIdx.x;
    if (i < n) out[i] = f2bf(in[i]);
}
// in[R][Cc] f32 -> out[Cc][R] bf16
__global__ void k_transpose_bf16(const float* __restrict__ in, __bf16* __restrict__ out,
                                 int R, int Cc) {
    int i = blockIdx.x * blockDim.x + threadIdx.x;
    if (i >= R * Cc) return;
    int r = i / Cc, c = i % Cc;
    out[(long)c * R + r] = f2bf(in[(long)r * Cc + c]);
}

// Depthwise causal conv over xBC + SiLU; also extracts bf16 B/C matrices.
__global__ void k_conv_silu(const float* __restrict__ zxbcdt,
                            const float* __restrict__ conv_w,
                            const float* __restrict__ conv_b,
                            float* __restrict__ xbc,
                            __bf16* __restrict__ Bbf, __bf16* __restrict__ Cbf)
{
    long i = (long)blockIdx.x * blockDim.x + threadIdx.x;
    if (i >= (long)TTOT * CONVD) return;
    int t  = (int)(i / CONVD);
    int ch = (int)(i % CONVD);
    int b  = t / LSEQ, lt = t % LSEQ;
    float acc = conv_b[ch];
#pragma unroll
    for (int k = 0; k < DCONV; ++k) {
        int tl = lt - (DCONV - 1) + k;
        if (tl >= 0) {
            float v = zxbcdt[(long)(b * LSEQ + tl) * DIP + DI + ch];
            acc += conv_w[ch * DCONV + k] * v;
        }
    }
    float s = siluf(acc);
    xbc[(long)t * CONVD + ch] = s;
    if (ch >= DI) {
        int n = ch - DI;
        if (n < DSTATE) Bbf[(long)t * DSTATE + n] = f2bf(s);
        else            Cbf[(long)t * DSTATE + (n - DSTATE)] = f2bf(s);
    }
}

__global__ void k_dt_softplus(const float* __restrict__ zxbcdt,
                              const float* __restrict__ dt_bias,
                              float* __restrict__ dt_sp)
{
    int i = blockIdx.x * blockDim.x + threadIdx.x;
    if (i >= TTOT * NH) return;
    int t = i / NH, h = i % NH;
    float x = zxbcdt[(long)t * DIP + 2 * DI + 2 * DSTATE + h] + dt_bias[h];
    float sp = (x > 20.0f) ? x : log1pf(__expf(x));
    dt_sp[t * NH + h] = sp;
}

// Per (b,c,h): within-chunk cumsum of dt*A. cs index g = ((b*NC+c)*NH+h)*CK + l.
__global__ void k_cumsum(const float* __restrict__ dt_sp, const float* __restrict__ A_log,
                         float* __restrict__ cs, float* __restrict__ Tch)
{
    int g = blockIdx.x * blockDim.x + threadIdx.x;
    if (g >= BATCH * NC * NH) return;
    int b = g / (NC * NH), c = (g / NH) % NC, h = g % NH;
    float Ah = -__expf(A_log[h]);
    float run = 0.0f;
    long base = (long)g * CK;
    for (int l = 0; l < CK; ++l) {
        int t = (b * NC + c) * CK + l;
        run += dt_sp[t * NH + h] * Ah;
        cs[base + l] = run;
    }
    Tch[(b * NH + h) * NC + c] = run;
}

// Per (b,c,h): BdT[n][l] = bf16(B[l,n]*exp(cs_last-cs[l])); XdT[p][l] = bf16(xh*dt)
__global__ __launch_bounds__(256)
void k_prep(const float* __restrict__ xbc, const float* __restrict__ dt_sp,
            const float* __restrict__ cs, __bf16* __restrict__ BdT,
            __bf16* __restrict__ XdT)
{
    int bch = blockIdx.x;                  // ((b*NC+c)*NH+h)
    int h = bch % NH;
    int bc = bch / NH;                     // b*NC+c
    const float* csp = cs + (long)bch * CK;
    float cl = csp[CK - 1];
    __bf16* bdt = BdT + (long)bch * DSTATE * CK;
    __bf16* xdt = XdT + (long)bch * HD * CK;
    for (int idx = threadIdx.x; idx < DSTATE * CK; idx += blockDim.x) {
        int n = idx / CK, l = idx % CK;
        int t = bc * CK + l;
        float v = xbc[(long)t * CONVD + DI + n] * __expf(cl - csp[l]);
        bdt[idx] = f2bf(v);
    }
    for (int idx = threadIdx.x; idx < HD * CK; idx += blockDim.x) {
        int p = idx / CK, l = idx % CK;
        int t = bc * CK + l;
        float v = xbc[(long)t * CONVD + h * HD + p] * dt_sp[t * NH + h];
        xdt[idx] = f2bf(v);
    }
}

// Inter-chunk scan: P_0=0; P_{z+1}=exp(T_z)*P_z + S_z. Emits PbfT[p][n] per chunk.
__global__ __launch_bounds__(256)
void k_scan(const float* __restrict__ states, const float* __restrict__ Tch,
            __bf16* __restrict__ PbfT)
{
    int bh = blockIdx.x;                   // b*NH+h
    int b = bh / NH, h = bh % NH;
    float P[32];
#pragma unroll
    for (int j = 0; j < 32; ++j) P[j] = 0.0f;
    for (int z = 0; z < NC; ++z) {
        long bch = (long)((b * NC + z) * NH + h);
#pragma unroll
        for (int j = 0; j < 32; ++j) {
            int e = threadIdx.x + j * 256;
            int n = e >> 6, p = e & 63;
            PbfT[bch * (HD * DSTATE) + p * DSTATE + n] = f2bf(P[j]);
        }
        if (z < NC - 1) {
            float dec = __expf(Tch[(b * NH + h) * NC + z]);
#pragma unroll
            for (int j = 0; j < 32; ++j) {
                int e = threadIdx.x + j * 256;
                P[j] = dec * P[j] + states[bch * (DSTATE * HD) + e];
            }
        }
    }
}

// Fused Y = (G .* expmask) @ Xd  +  diag(exp(cs)) * (C @ P^T)  +  D*xh
// One block (512 thr = 16 waves) per (b,c,h); each wave owns 16 rows x 64 cols.
__global__ __launch_bounds__(512)
void k_ydiag(const float* __restrict__ G, const float* __restrict__ cs,
             const __bf16* __restrict__ XdT, const __bf16* __restrict__ Cbf,
             const __bf16* __restrict__ PbfT, const float* __restrict__ xbc,
             const float* __restrict__ Dvec, float* __restrict__ Y)
{
    __shared__ __bf16 Ms[CK][64];          // 32 KB staged masked-decay tile
    __shared__ float  cs_sh[CK];

    const int bch = blockIdx.x;
    const int h   = bch % NH;
    const int bc  = bch / NH;
    const int lane = threadIdx.x & 31;
    const int wid  = threadIdx.x >> 5;
    const int m0   = wid * 16;             // 16 waves cover 256 rows

    const float* csp = cs + (long)bch * CK;
    const float* Gp  = G + (long)bc * CK * CK;
    const __bf16* xdt = XdT + (long)bch * HD * CK;
    const __bf16* pbt = PbfT + (long)bch * HD * DSTATE;

    if (threadIdx.x < CK) cs_sh[threadIdx.x] = csp[threadIdx.x];
    __syncthreads();

    v8f acc[4], aco[4];
#pragma unroll
    for (int nt = 0; nt < 4; ++nt) {
        acc[nt] = (v8f){0,0,0,0,0,0,0,0};
        aco[nt] = (v8f){0,0,0,0,0,0,0,0};
    }

    // --- diagonal part: K = s, staged 64 wide at a time ---
    for (int kt = 0; kt < 4; ++kt) {
        __syncthreads();
        for (int idx = threadIdx.x; idx < CK * 64; idx += 512) {
            int l = idx >> 6, so = idx & 63;
            int s = kt * 64 + so;
            float v = 0.0f;
            if (l >= s) v = Gp[(long)l * CK + s] * __expf(cs_sh[l] - cs_sh[s]);
            Ms[l][so] = f2bf(v);
        }
        __syncthreads();
#pragma unroll
        for (int k2 = 0; k2 < 64; k2 += 32) {
            const int klo = (lane < 16) ? 0 : 8;
            const __bf16* mr = &Ms[m0 + (lane & 15)][k2];
            union { v16bf v; v8bf h2[2]; } ua;
            ua.h2[0] = *(const v8bf*)(mr + klo);
            ua.h2[1] = *(const v8bf*)(mr + klo + 16);
            const int s_off = kt * 64 + k2;
#pragma unroll
            for (int nt = 0; nt < 4; ++nt) {
                const int p = nt * 16 + (lane & 15);
                v16bf bfrag = load_b_frag(xdt + (long)p * CK + s_off, lane);
                acc[nt] = __builtin_amdgcn_wmma_f32_16x16x32_bf16(
                    false, ua.v, false, bfrag, (short)0, acc[nt], false, false);
            }
        }
    }

    // --- off-diagonal part: K = n (128) ---
    const __bf16* crow = Cbf + (long)(bc * CK + m0 + (lane & 15)) * DSTATE;
#pragma unroll
    for (int k = 0; k < DSTATE; k += 32) {
        v16bf a = load_a_frag(crow + k, lane);
#pragma unroll
        for (int nt = 0; nt < 4; ++nt) {
            const int p = nt * 16 + (lane & 15);
            v16bf b = load_b_frag(pbt + (long)p * DSTATE + k, lane);
            aco[nt] = __builtin_amdgcn_wmma_f32_16x16x32_bf16(
                false, a, false, b, (short)0, aco[nt], false, false);
        }
    }

    // --- epilogue: combine + D skip ---
    const float Dh = Dvec[h];
#pragma unroll
    for (int nt = 0; nt < 4; ++nt) {
#pragma unroll
        for (int r = 0; r < 8; ++r) {
            const int l = m0 + r + ((lane < 16) ? 0 : 8);
            const int p = nt * 16 + (lane & 15);
            const int t = bc * CK + l;
            float xh = xbc[(long)t * CONVD + h * HD + p];
            float y = acc[nt][r] + __expf(cs_sh[l]) * aco[nt][r] + Dh * xh;
            Y[(long)t * DI + h * HD + p] = y;
        }
    }
}

// Gate (y * silu(z)) + RMSNorm -> bf16 row for the output GEMM.
__global__ __launch_bounds__(256)
void k_gate_norm(const float* __restrict__ Y, const float* __restrict__ zxbcdt,
                 const float* __restrict__ norm_w, __bf16* __restrict__ ybf)
{
    __shared__ float red[256];
    const int t = blockIdx.x;
    float vals[16];
    float ss = 0.0f;
#pragma unroll
    for (int i = 0; i < 16; ++i) {
        int e = threadIdx.x + i * 256;
        float z = zxbcdt[(long)t * DIP + e];
        float v = Y[(long)t * DI + e] * siluf(z);
        vals[i] = v;
        ss += v * v;
    }
    red[threadIdx.x] = ss;
    __syncthreads();
    for (int s = 128; s > 0; s >>= 1) {
        if (threadIdx.x < s) red[threadIdx.x] += red[threadIdx.x + s];
        __syncthreads();
    }
    float scale = rsqrtf(red[0] / (float)DI + 1e-5f);
#pragma unroll
    for (int i = 0; i < 16; ++i) {
        int e = threadIdx.x + i * 256;
        ybf[(long)t * DI + e] = f2bf(vals[i] * scale * norm_w[e]);
    }
}

// ------------------------------ Host launcher ------------------------------
extern "C" void kernel_launch(void* const* d_in, const int* in_sizes, int n_in,
                              void* d_out, int out_size, void* d_ws, size_t ws_size,
                              hipStream_t stream)
{
    const float* x       = (const float*)d_in[0];
    const float* w_in    = (const float*)d_in[1];
    const float* conv_w  = (const float*)d_in[2];
    const float* conv_b  = (const float*)d_in[3];
    const float* dt_bias = (const float*)d_in[4];
    const float* A_log   = (const float*)d_in[5];
    const float* Dvec    = (const float*)d_in[6];
    const float* norm_w  = (const float*)d_in[7];
    const float* w_out   = (const float*)d_in[8];
    float* out = (float*)d_out;

    // workspace carving (256B aligned)
    char* p = (char*)d_ws;
    auto alloc = [&](size_t bytes) -> void* {
        void* r = (void*)p;
        p += (bytes + 255) & ~(size_t)255;
        return r;
    };
    __bf16* x_bf    = (__bf16*)alloc((size_t)TTOT * DM * 2);
    __bf16* w_inT   = (__bf16*)alloc((size_t)DIP * DM * 2);
    __bf16* w_outT  = (__bf16*)alloc((size_t)DM * DI * 2);
    float*  zxbcdt  = (float*)alloc((size_t)TTOT * DIP * 4);
    float*  xbc     = (float*)alloc((size_t)TTOT * CONVD * 4);
    float*  dt_sp   = (float*)alloc((size_t)TTOT * NH * 4);
    float*  cs      = (float*)alloc((size_t)BATCH * NC * NH * CK * 4);
    float*  Tch     = (float*)alloc((size_t)BATCH * NH * NC * 4);
    __bf16* Bbf     = (__bf16*)alloc((size_t)TTOT * DSTATE * 2);
    __bf16* Cbf     = (__bf16*)alloc((size_t)TTOT * DSTATE * 2);
    float*  G       = (float*)alloc((size_t)BATCH * NC * CK * CK * 4);
    __bf16* BdT     = (__bf16*)alloc((size_t)BATCH * NC * NH * DSTATE * CK * 2);
    __bf16* XdT     = (__bf16*)alloc((size_t)BATCH * NC * NH * HD * CK * 2);
    float*  states  = (float*)alloc((size_t)BATCH * NC * NH * DSTATE * HD * 4);
    __bf16* PbfT    = (__bf16*)alloc((size_t)BATCH * NC * NH * HD * DSTATE * 2);
    float*  Y       = (float*)alloc((size_t)TTOT * DI * 4);
    __bf16* ybf     = (__bf16*)alloc((size_t)TTOT * DI * 2);

    // 1. operand conversion / transposes
    {
        int n = TTOT * DM;
        k_cvt_bf16<<<(n + 255) / 256, 256, 0, stream>>>(x, x_bf, n);
    }
    k_transpose_bf16<<<((long)DM * DIP + 255) / 256, 256, 0, stream>>>(w_in, w_inT, DM, DIP);
    k_transpose_bf16<<<((long)DI * DM + 255) / 256, 256, 0, stream>>>(w_out, w_outT, DI, DM);

    // 2. in_proj GEMM: zxbcdt[4096,8512] = x_bf @ w_inT
    k_gemm<0><<<dim3(DIP / 64, TTOT / 128, 1), 256, 0, stream>>>(
        x_bf, w_inT, zxbcdt, nullptr, TTOT, DIP, DM, DM, DM, DIP, 0, 0, 0);

    // 3. conv + silu (+ B/C extraction), softplus(dt), cumsum
    k_conv_silu<<<((long)TTOT * CONVD + 255) / 256, 256, 0, stream>>>(
        zxbcdt, conv_w, conv_b, xbc, Bbf, Cbf);
    k_dt_softplus<<<(TTOT * NH + 255) / 256, 256, 0, stream>>>(zxbcdt, dt_bias, dt_sp);
    k_cumsum<<<(BATCH * NC * NH + 255) / 256, 256, 0, stream>>>(dt_sp, A_log, cs, Tch);

    // 4. G = C @ B^T per (b,c): batched WMMA GEMM (M=256,N=256,K=128)
    k_gemm<0><<<dim3(CK / 64, CK / 128, BATCH * NC), 256, 0, stream>>>(
        Cbf, Bbf, G, nullptr, CK, CK, DSTATE, DSTATE, DSTATE, CK,
        (long)CK * DSTATE, (long)CK * DSTATE, (long)CK * CK);

    // 5. decay-scaled transposed operands per (b,c,h)
    k_prep<<<BATCH * NC * NH, 256, 0, stream>>>(xbc, dt_sp, cs, BdT, XdT);

    // 6. chunk states per (b,c,h): [128,64] = BdT @ XdT^T (M=128,N=64,K=256)
    k_gemm<0><<<dim3(1, 1, BATCH * NC * NH), 256, 0, stream>>>(
        BdT, XdT, states, nullptr, DSTATE, HD, CK, CK, CK, HD,
        (long)DSTATE * CK, (long)HD * CK, (long)DSTATE * HD);

    // 7. inter-chunk scan -> prefix states (bf16, transposed)
    k_scan<<<BATCH * NH, 256, 0, stream>>>(states, Tch, PbfT);

    // 8. fused Y_diag + Y_off + D skip
    k_ydiag<<<BATCH * NC * NH, 512, 0, stream>>>(G, cs, XdT, Cbf, PbfT, xbc, Dvec, Y);

    // 9. gate + RMSNorm -> bf16
    k_gate_norm<<<TTOT, 256, 0, stream>>>(Y, zxbcdt, norm_w, ybf);

    // 10. out_proj GEMM with residual: out = x + ybf @ w_outT
    k_gemm<1><<<dim3(DM / 64, TTOT / 128, 1), 256, 0, stream>>>(
        ybf, w_outT, out, x, TTOT, DM, DI, DI, DI, DM, 0, 0, 0);
}
